// WindowAttention_21586505630111
// MI455X (gfx1250) — matrix operand order repaired and checked by reading, not compiled, not run on hardware
//
#include <hip/hip_runtime.h>
#include <hip/hip_bf16.h>

// ---- problem constants (from reference) ----
#define WSZ   8
#define NTOK  64      // WS*WS
#define CDIM  256
#define NHEAD 8
#define HDIM  32
#define NWIN  1024

typedef __attribute__((ext_vector_type(16))) _Float16 v16h;
typedef __attribute__((ext_vector_type(8)))  _Float16 v8h;
typedef __attribute__((ext_vector_type(4)))  _Float16 v4h;
typedef __attribute__((ext_vector_type(8)))  float    v8f;

// ---- LDS layout (dynamic shared) ----
constexpr int LDX  = CDIM + 8;         // 264 f16/row (16B-aligned rows)
constexpr int LDH  = HDIM + 8;         // 40 f16/row
constexpr int LDP  = 64 + 8;           // 72 f16/row (P staging)
constexpr int LDVT = 64 + 8;           // 72 f16/row (transposed V: 32 rows x 64 cols)
constexpr int SX_ELEMS     = NTOK * LDX;
constexpr int REGION_ELEMS = NTOK * LDH;               // 2560 f16 (Vt needs 32*72=2304, fits)
constexpr int HEADS_ELEMS  = NHEAD * 2 * REGION_ELEMS; // per head: [Q->Vt][K->O]
constexpr int SP_ELEMS     = NHEAD * 16 * LDP;
constexpr int BIAS_FLOATS  = (2 * WSZ - 1) * (2 * WSZ - 1) * NHEAD;
constexpr int MASK_FLOATS  = NTOK * NTOK;
constexpr size_t SMEM_BYTES =
    (size_t)(SX_ELEMS + HEADS_ELEMS + SP_ELEMS) * sizeof(_Float16)
  + (size_t)(BIAS_FLOATS + MASK_FLOATS) * sizeof(float);             // ~157.7 KB

constexpr int NQ_W = 3 * CDIM * CDIM;   // w_qkv elements
constexpr int NP_W = CDIM * CDIM;       // w_proj elements
constexpr size_t WS_NEEDED = (size_t)(NQ_W + NP_W) * sizeof(_Float16);  // 512 KB

__device__ __forceinline__ v8f wmma16(v16h a, v16h b, v8f c) {
  // D(16x16,f32) = A(16x32,f16) * B(32x16,f16) + C
  return __builtin_amdgcn_wmma_f32_16x16x32_f16(false, a, false, b,
                                                (short)0, c, false, false);
}

// A fragment: lane holds row m0+(lane%16); K pattern (hf*16 + 0..15)
__device__ __forceinline__ v16h fragA_lds(const _Float16* p, int ld, int m0,
                                          int k0, int l, int hf) {
  const _Float16* r = p + (m0 + l) * ld + k0 + hf * 16;
  v16h f;
#pragma unroll
  for (int i = 0; i < 8; ++i) { f[i] = r[i]; f[i + 8] = r[i + 8]; }
  return f;
}

// B fragment where B's column n is a contiguous row of a row-major LDS matrix
__device__ __forceinline__ v16h fragB_rows_lds(const _Float16* p, int ld, int n0,
                                               int k0, int l, int hf) {
  const _Float16* r = p + (n0 + l) * ld + k0 + hf * 16;
  v16h f;
#pragma unroll
  for (int i = 0; i < 16; ++i) f[i] = r[i];
  return f;
}

// Same, sourced from global weights (f32 or pre-converted f16)
template <typename WT>
__device__ __forceinline__ v16h fragB_rows_gw(const WT* __restrict__ p, int ld,
                                              int n0, int k0, int l, int hf) {
  const WT* r = p + (size_t)(n0 + l) * ld + k0 + hf * 16;
  v16h f;
#pragma unroll
  for (int i = 0; i < 16; ++i) f[i] = (_Float16)r[i];
  return f;
}

__device__ __forceinline__ float redmax16(float v) {
  v = fmaxf(v, __shfl_xor(v, 1, 32));
  v = fmaxf(v, __shfl_xor(v, 2, 32));
  v = fmaxf(v, __shfl_xor(v, 4, 32));
  v = fmaxf(v, __shfl_xor(v, 8, 32));
  return v;
}
__device__ __forceinline__ float redsum16(float v) {
  v += __shfl_xor(v, 1, 32);
  v += __shfl_xor(v, 2, 32);
  v += __shfl_xor(v, 4, 32);
  v += __shfl_xor(v, 8, 32);
  return v;
}

// 64x32 GEMM: sX(64x256) * W[rowBase..rowBase+32)^T -> f16 LDS.
// TSTORE=false: row-major 64x32 (stride LDH). TSTORE=true: transposed 32x64 (stride LDVT),
// which makes the D-fragment stores contiguous (v8h = ds_store_b128).
template <typename WT, bool TSTORE>
__device__ __forceinline__ void gemm_qkv(const _Float16* sX,
                                         const WT* __restrict__ W, int rowBase,
                                         _Float16* dst, float scale, int l, int hf) {
  v8f acc[4][2];
#pragma unroll
  for (int tm = 0; tm < 4; ++tm)
#pragma unroll
    for (int tn = 0; tn < 2; ++tn)
#pragma unroll
      for (int r = 0; r < 8; ++r) acc[tm][tn][r] = 0.0f;

#pragma unroll
  for (int ks = 0; ks < 8; ++ks) {
    v16h b0 = fragB_rows_gw<WT>(W, CDIM, rowBase,      ks * 32, l, hf);
    v16h b1 = fragB_rows_gw<WT>(W, CDIM, rowBase + 16, ks * 32, l, hf);
#pragma unroll
    for (int tm = 0; tm < 4; ++tm) {
      v16h a = fragA_lds(sX, LDX, tm * 16, ks * 32, l, hf);
      acc[tm][0] = wmma16(a, b0, acc[tm][0]);
      acc[tm][1] = wmma16(a, b1, acc[tm][1]);
    }
  }
#pragma unroll
  for (int tm = 0; tm < 4; ++tm)
#pragma unroll
    for (int tn = 0; tn < 2; ++tn) {
      if (TSTORE) {
        v8h pk;
#pragma unroll
        for (int r = 0; r < 8; ++r) pk[r] = (_Float16)(acc[tm][tn][r] * scale);
        *(v8h*)(dst + (tn * 16 + l) * LDVT + tm * 16 + hf * 8) = pk;
      } else {
#pragma unroll
        for (int r = 0; r < 8; ++r) {
          int row = tm * 16 + hf * 8 + r;
          dst[row * LDH + tn * 16 + l] = (_Float16)(acc[tm][tn][r] * scale);
        }
      }
    }
}

template <typename WT>
__global__ __launch_bounds__(256, 1)
void win_attn_fused(const float* __restrict__ x,
                    const WT* __restrict__ w_qkv,
                    const WT* __restrict__ w_proj,
                    const float* __restrict__ b_proj,
                    const float* __restrict__ bias_table,
                    const float* __restrict__ mask,
                    float* __restrict__ out) {
  extern __shared__ __align__(16) char smem[];
  _Float16* sX     = (_Float16*)smem;
  _Float16* sHeads = sX + SX_ELEMS;
  _Float16* sP     = sHeads + HEADS_ELEMS;
  float*    sBias  = (float*)(sP + SP_ELEMS);
  float*    sMask  = sBias + BIAS_FLOATS;

  const int tid  = threadIdx.x;
  const int lane = tid & 31;
  const int wave = tid >> 5;     // == head index
  const int l    = lane & 15;
  const int hf   = lane >> 4;
  const int b    = blockIdx.x;
  const int h    = wave;

  // -------- phase 0: stage x (f32->f16), mask, bias into LDS --------
  {
    const float4* x4 = (const float4*)(x + (size_t)b * NTOK * CDIM);
    for (int t = tid; t < NTOK * CDIM / 4; t += 256) {
      float4 v = x4[t];
      int row = t >> 6;
      int c   = (t & 63) << 2;
      v4h p;
      p[0] = (_Float16)v.x; p[1] = (_Float16)v.y;
      p[2] = (_Float16)v.z; p[3] = (_Float16)v.w;
      *(v4h*)(sX + row * LDX + c) = p;
    }
    const float4* m4 = (const float4*)(mask + (size_t)(b % NWIN) * NTOK * NTOK);
    float4* sm4 = (float4*)sMask;
    for (int t = tid; t < MASK_FLOATS / 4; t += 256) sm4[t] = m4[t];
    for (int t = tid; t < BIAS_FLOATS; t += 256) sBias[t] = bias_table[t];
  }
  __syncthreads();

  _Float16* regQV = sHeads + h * 2 * REGION_ELEMS;  // Q, later transposed V
  _Float16* regKO = regQV + REGION_ELEMS;           // K, later attn-out
  _Float16* sPw   = sP + wave * 16 * LDP;

  // -------- phase 1: per-head Q,K projections (weights streamed from L2) -----
  const float scale = 0.17677669529663687f;         // hd^-0.5
  gemm_qkv<WT, false>(sX, w_qkv, h * HDIM,        regQV, scale, l, hf);
  gemm_qkv<WT, false>(sX, w_qkv, CDIM + h * HDIM, regKO, 1.0f,  l, hf);

  // -------- phase 2: S = Q*K^T + bias + mask (C operand carries bias+mask) ---
  v8f sacc[4][4];
#pragma unroll
  for (int tm = 0; tm < 4; ++tm) {
    v16h aq = fragA_lds(regQV, LDH, tm * 16, 0, l, hf);
#pragma unroll
    for (int tn = 0; tn < 4; ++tn) {
      const int j = tn * 16 + l;
      const int jy = j >> 3, jx = j & 7;
      v8f c;
#pragma unroll
      for (int r = 0; r < 8; ++r) {
        int i = tm * 16 + hf * 8 + r;
        int iy = i >> 3, ix = i & 7;
        int rp = (iy - jy + 7) * (2 * WSZ - 1) + (ix - jx + 7);
        c[r] = sBias[rp * NHEAD + h] + sMask[i * NTOK + j];
      }
      v16h bk = fragB_rows_lds(regKO, LDH, tn * 16, 0, l, hf);
      sacc[tm][tn] = wmma16(aq, bk, c);
    }
  }

  // -------- phase 3: V projection, stored TRANSPOSED (overwrites Q region) ---
  gemm_qkv<WT, true>(sX, w_qkv, 2 * CDIM + h * HDIM, regQV, 1.0f, l, hf);

  // -------- phase 4: softmax (register + shfl) and O = P*V, per 16-row tile --
#pragma unroll
  for (int tm = 0; tm < 4; ++tm) {
    float inv[8];
#pragma unroll
    for (int r = 0; r < 8; ++r) {
      float m = fmaxf(fmaxf(sacc[tm][0][r], sacc[tm][1][r]),
                      fmaxf(sacc[tm][2][r], sacc[tm][3][r]));
      m = redmax16(m);
      float s = 0.0f;
#pragma unroll
      for (int tn = 0; tn < 4; ++tn) {
        float p = __expf(sacc[tm][tn][r] - m);
        sacc[tm][tn][r] = p;
        s += p;
      }
      s = redsum16(s);
      inv[r] = 1.0f / s;
    }
#pragma unroll
    for (int tn = 0; tn < 4; ++tn)
#pragma unroll
      for (int r = 0; r < 8; ++r)
        sPw[(hf * 8 + r) * LDP + tn * 16 + l] =
            (_Float16)(sacc[tm][tn][r] * inv[r]);

    v8f o0, o1;
#pragma unroll
    for (int r = 0; r < 8; ++r) { o0[r] = 0.0f; o1[r] = 0.0f; }
#pragma unroll
    for (int ks = 0; ks < 2; ++ks) {
      v16h ap = fragA_lds(sPw, LDP, 0, ks * 32, l, hf);
      // Vt: row = channel, col = token -> B-fragment is 16 contiguous f16
      v16h bv0 = fragB_rows_lds(regQV, LDVT, 0,  ks * 32, l, hf);
      v16h bv1 = fragB_rows_lds(regQV, LDVT, 16, ks * 32, l, hf);
      o0 = wmma16(ap, bv0, o0);
      o1 = wmma16(ap, bv1, o1);
    }
#pragma unroll
    for (int r = 0; r < 8; ++r) {
      int row = tm * 16 + hf * 8 + r;
      regKO[row * LDH + l]      = (_Float16)o0[r];   // overwrite K region
      regKO[row * LDH + 16 + l] = (_Float16)o1[r];
    }
  }
  __syncthreads();  // all heads' outputs visible

  // -------- phase 5: out = O(64x256) * w_proj^T + b_proj -------------------
  {
    const int n0 = wave * HDIM;
    const float bp0 = b_proj[n0 + l];
    const float bp1 = b_proj[n0 + 16 + l];
    v8f acc[4][2];
#pragma unroll
    for (int tm = 0; tm < 4; ++tm)
#pragma unroll
      for (int r = 0; r < 8; ++r) { acc[tm][0][r] = bp0; acc[tm][1][r] = bp1; }

#pragma unroll
    for (int kk = 0; kk < 8; ++kk) {     // channel block == head kk
      const _Float16* regO = sHeads + kk * 2 * REGION_ELEMS + REGION_ELEMS;
      v16h b0 = fragB_rows_gw<WT>(w_proj, CDIM, n0,      kk * 32, l, hf);
      v16h b1 = fragB_rows_gw<WT>(w_proj, CDIM, n0 + 16, kk * 32, l, hf);
#pragma unroll
      for (int tm = 0; tm < 4; ++tm) {
        v16h a = fragA_lds(regO, LDH, tm * 16, 0, l, hf);
        acc[tm][0] = wmma16(a, b0, acc[tm][0]);
        acc[tm][1] = wmma16(a, b1, acc[tm][1]);
      }
    }
    float* op = out + (size_t)b * NTOK * CDIM;
#pragma unroll
    for (int tm = 0; tm < 4; ++tm)
#pragma unroll
      for (int r = 0; r < 8; ++r) {
        int row = tm * 16 + hf * 8 + r;
        op[row * CDIM + n0 + l]      = acc[tm][0][r];
        op[row * CDIM + n0 + 16 + l] = acc[tm][1][r];
      }
  }
}

// One-shot weight conversion f32 -> f16 into workspace (weights then L2-resident as f16)
__global__ __launch_bounds__(256)
void cvt_weights(const float* __restrict__ wq, const float* __restrict__ wp,
                 _Float16* __restrict__ dst) {
  for (int t = blockIdx.x * blockDim.x + threadIdx.x; t < NQ_W + NP_W;
       t += gridDim.x * blockDim.x) {
    float v = (t < NQ_W) ? wq[t] : wp[t - NQ_W];
    dst[t] = (_Float16)v;
  }
}

extern "C" void kernel_launch(void* const* d_in, const int* in_sizes, int n_in,
                              void* d_out, int out_size, void* d_ws, size_t ws_size,
                              hipStream_t stream) {
  const float* x          = (const float*)d_in[0];
  const float* w_qkv      = (const float*)d_in[1];
  const float* w_proj     = (const float*)d_in[2];
  const float* b_proj     = (const float*)d_in[3];
  const float* bias_table = (const float*)d_in[4];
  const float* mask       = (const float*)d_in[5];
  float* out = (float*)d_out;
  (void)n_in; (void)out_size;

  const int nBlocks = in_sizes[0] / (NTOK * CDIM);   // B_ = 8192

  if (ws_size >= WS_NEEDED && d_ws != nullptr) {
    _Float16* w16q = (_Float16*)d_ws;
    _Float16* w16p = w16q + NQ_W;
    cvt_weights<<<512, 256, 0, stream>>>(w_qkv, w_proj, w16q);
    (void)hipFuncSetAttribute(
        reinterpret_cast<const void*>(&win_attn_fused<_Float16>),
        hipFuncAttributeMaxDynamicSharedMemorySize, (int)SMEM_BYTES);
    win_attn_fused<_Float16><<<nBlocks, 256, SMEM_BYTES, stream>>>(
        x, w16q, w16p, b_proj, bias_table, mask, out);
  } else {
    (void)hipFuncSetAttribute(
        reinterpret_cast<const void*>(&win_attn_fused<float>),
        hipFuncAttributeMaxDynamicSharedMemorySize, (int)SMEM_BYTES);
    win_attn_fused<float><<<nBlocks, 256, SMEM_BYTES, stream>>>(
        x, w_qkv, w_proj, b_proj, bias_table, mask, out);
  }
}
